// Wav2Vec2GumbelVectorQuantizer_34909494181920
// MI455X (gfx1250) — compile-verified, matrix operand order checked by reading, not run on hardware
//
#include <hip/hip_runtime.h>
#include <cstdint>
#include <cstddef>

// ---------------------------------------------------------------------------
// Wav2Vec2 Gumbel VQ forward for MI455X (gfx1250, wave32, WMMA)
//   B=16, S=4096, H=512, G=2, V=320, D/G=128, BT=65536, N=G*V=640
// GEMM via v_wmma_f32_16x16x32_bf16 (f32 accumulate), memory-bound by design.
// v2: each wave register-blocks TWO 16-row M sub-tiles so every W (B-matrix)
//     fragment fetched from L2 feeds two WMMAs -> halves L2 W-traffic.
// ---------------------------------------------------------------------------

typedef __attribute__((ext_vector_type(16))) __bf16 v16bf;
typedef __attribute__((ext_vector_type(8)))  float  v8f;

union Frag {
  uint4  u[2];
  v16bf  v;
};

__device__ __forceinline__ unsigned short f2bf(float f) {
  unsigned x = __float_as_uint(f);
  x += 0x7fffu + ((x >> 16) & 1u);   // round-to-nearest-even
  return (unsigned short)(x >> 16);
}

// ---------------------------------------------------------------------------
// Kernel 1: W (512x640 f32, row-major K x N) -> Wt (640x512 bf16, N-major)
//           and zero the 640-bin histogram.
// ---------------------------------------------------------------------------
__global__ void vq_prep(const float* __restrict__ W,
                        unsigned short* __restrict__ Wt,
                        unsigned* __restrict__ hist) {
  const int n = blockIdx.x;            // 0..639
  if (threadIdx.x == 0) hist[n] = 0u;
  for (int k = threadIdx.x; k < 512; k += blockDim.x) {
    Wt[(size_t)n * 512 + k] = f2bf(W[(size_t)k * 640 + n]);
  }
}

// ---------------------------------------------------------------------------
// Kernel 2: main fused GEMM + argmax + gather + histogram.
// Block: 256 threads (8 waves). Block tile: 128 tokens.
// Wave w: rows (w&3)*32 .. +31 (two 16-row sub-tiles), group g = w>>2.
// ---------------------------------------------------------------------------
__global__ __launch_bounds__(256)
void vq_main(const float* __restrict__ hidden,
             const unsigned short* __restrict__ Wt,
             const float* __restrict__ bias,
             const float* __restrict__ codev,
             float* __restrict__ out,
             unsigned* __restrict__ hist) {
  extern __shared__ unsigned short lds16[];   // 128 rows x 512 bf16, swizzled
  const int tid = threadIdx.x;
  const int tokBase = blockIdx.x << 7;        // 128 tokens per block

  // ---- Phase 1: stage hidden tile (128x512 f32) -> bf16 LDS ----------------
  // XOR swizzle on 8-element (16B) chunks: chunk' = chunk ^ (row & 15).
  {
    const float* hsrc = hidden + (size_t)tokBase * 512;
    for (int i = tid; i < 128 * 128; i += 256) {    // 128 float4 per row
      const int r = i >> 7;
      const int k = (i & 127) << 2;                 // col, multiple of 4
      const float4 v = *(const float4*)(hsrc + (size_t)r * 512 + k);
      const unsigned lo = (unsigned)f2bf(v.x) | ((unsigned)f2bf(v.y) << 16);
      const unsigned hi = (unsigned)f2bf(v.z) | ((unsigned)f2bf(v.w) << 16);
      const unsigned sidx =
          (unsigned)(r * 512 + (((k >> 3) ^ (r & 15)) << 3) + (k & 7));
      *(uint2*)&lds16[sidx] = make_uint2(lo, hi);
    }
  }
  __syncthreads();

  const int wave = tid >> 5;
  const int lane = tid & 31;
  const int row  = lane & 15;    // M for A/C fragments, N for B fragments
  const int half = lane >> 4;    // selects K sub-block per ISA 16-bit layout
  const int g    = wave >> 2;    // group 0/1
  const int m0   = (wave & 3) << 5;   // 32 rows per wave
  const int arow0 = m0 + row;
  const int arow1 = m0 + 16 + row;

  // ---- Hoist A fragments for BOTH 16-row sub-tiles (K = 0..511) ------------
  // 16-bit A 16x32 layout: lane(M=row, half): VGPR0-3 = K[half*8..+7],
  //                                           VGPR4-7 = K[16+half*8..+7].
  Frag afr0[16], afr1[16];
#pragma unroll
  for (int kt = 0; kt < 16; ++kt) {
    const int kb = (kt << 5) + (half << 3);
    const unsigned d0 = (unsigned)((((kb)      >> 3)) << 3);
    const unsigned d1 = (unsigned)((((kb + 16) >> 3)) << 3);
    const unsigned s0a = (unsigned)(((arow0 & 15) << 3));
    const unsigned s1a = (unsigned)(((arow1 & 15) << 3));
    afr0[kt].u[0] = *(const uint4*)&lds16[arow0 * 512 + (d0 ^ s0a)];
    afr0[kt].u[1] = *(const uint4*)&lds16[arow0 * 512 + (d1 ^ s0a)];
    afr1[kt].u[0] = *(const uint4*)&lds16[arow1 * 512 + (d0 ^ s1a)];
    afr1[kt].u[1] = *(const uint4*)&lds16[arow1 * 512 + (d1 ^ s1a)];
  }

  // Running argmax state per sub-tile: acc VGPR r = (M = r + 8*half, N = row).
  float rv0[8], rv1[8];
  int   ri0[8], ri1[8];
#pragma unroll
  for (int r = 0; r < 8; ++r) {
    rv0[r] = -3.4e38f; ri0[r] = 0;
    rv1[r] = -3.4e38f; ri1[r] = 0;
  }

  // ---- Phase 2: sweep this group's 320 logits in 20 N-tiles ----------------
  for (int t = 0; t < 20; ++t) {
    const int n0 = g * 320 + (t << 4);
    v8f acc0 = {0.f, 0.f, 0.f, 0.f, 0.f, 0.f, 0.f, 0.f};
    v8f acc1 = {0.f, 0.f, 0.f, 0.f, 0.f, 0.f, 0.f, 0.f};
    const unsigned short* wt =
        Wt + (size_t)(n0 + row) * 512 + (half << 3);
#pragma unroll
    for (int kt = 0; kt < 16; ++kt) {
      Frag bf;
      bf.u[0] = *(const uint4*)(wt + (kt << 5));
      bf.u[1] = *(const uint4*)(wt + (kt << 5) + 16);
      acc0 = __builtin_amdgcn_wmma_f32_16x16x32_bf16(
          false, afr0[kt].v, false, bf.v, (short)0, acc0, false, false);
      acc1 = __builtin_amdgcn_wmma_f32_16x16x32_bf16(
          false, afr1[kt].v, false, bf.v, (short)0, acc1, false, false);
    }
    const float bb = bias[n0 + row];
    const int colLocal = (t << 4) + row;   // index within group (0..319)
#pragma unroll
    for (int r = 0; r < 8; ++r) {
      const float v0 = acc0[r] + bb;
      const bool b0 = v0 > rv0[r];
      rv0[r] = b0 ? v0 : rv0[r];
      ri0[r] = b0 ? colLocal : ri0[r];
      const float v1 = acc1[r] + bb;
      const bool b1 = v1 > rv1[r];
      rv1[r] = b1 ? v1 : rv1[r];
      ri1[r] = b1 ? colLocal : ri1[r];
    }
  }

  // ---- Phase 3: argmax reduction across the 16 columns (16-lane halves) ----
#pragma unroll
  for (int r = 0; r < 8; ++r) {
#pragma unroll
    for (int m = 1; m < 16; m <<= 1) {
      {
        const float ov = __shfl_xor(rv0[r], m, 32);
        const int   oi = __shfl_xor(ri0[r], m, 32);
        const bool take = (ov > rv0[r]) || (ov == rv0[r] && oi < ri0[r]);
        rv0[r] = take ? ov : rv0[r];
        ri0[r] = take ? oi : ri0[r];
      }
      {
        const float ov = __shfl_xor(rv1[r], m, 32);
        const int   oi = __shfl_xor(ri1[r], m, 32);
        const bool take = (ov > rv1[r]) || (ov == rv1[r] && oi < ri1[r]);
        rv1[r] = take ? ov : rv1[r];
        ri1[r] = take ? oi : ri1[r];
      }
    }
  }

  // ---- Phase 4: codebook gather, output write, histogram -------------------
  // Half h owns rows h*8..h*8+7 of each sub-tile; 16 lanes copy 128 floats/row.
#pragma unroll
  for (int r = 0; r < 8; ++r) {
    {
      const int idx   = ri0[r];
      const int token = tokBase + m0 + (half << 3) + r;
      const float4* cv =
          (const float4*)(codev + ((size_t)g * 320 + (size_t)idx) * 128);
      float4* op = (float4*)(out + (size_t)token * 256 + g * 128);
      op[row]      = cv[row];
      op[row + 16] = cv[row + 16];
      if (row == 0) atomicAdd(&hist[g * 320 + idx], 1u);
    }
    {
      const int idx   = ri1[r];
      const int token = tokBase + m0 + 16 + (half << 3) + r;
      const float4* cv =
          (const float4*)(codev + ((size_t)g * 320 + (size_t)idx) * 128);
      float4* op = (float4*)(out + (size_t)token * 256 + g * 128);
      op[row]      = cv[row];
      op[row + 16] = cv[row + 16];
      if (row == 0) atomicAdd(&hist[g * 320 + idx], 1u);
    }
  }
}

// ---------------------------------------------------------------------------
// Kernel 3: histogram -> perplexity scalar at out[B*S*G*Dg] = out[16777216].
// ---------------------------------------------------------------------------
__global__ void vq_finalize(const unsigned* __restrict__ hist,
                            float* __restrict__ outp) {
  __shared__ float s0[256];
  __shared__ float s1[256];
  const int t = threadIdx.x;
  float h0 = 0.f, h1 = 0.f;
  for (int i = t; i < 320; i += 256) {
    const float p = (float)hist[i]       * (1.0f / 65536.0f);
    const float q = (float)hist[320 + i] * (1.0f / 65536.0f);
    h0 += p * logf(p + 1e-7f);
    h1 += q * logf(q + 1e-7f);
  }
  s0[t] = h0; s1[t] = h1;
  __syncthreads();
  for (int s = 128; s > 0; s >>= 1) {
    if (t < s) { s0[t] += s0[t + s]; s1[t] += s1[t + s]; }
    __syncthreads();
  }
  if (t == 0) *outp = expf(-s0[0]) + expf(-s1[0]);
}

// ---------------------------------------------------------------------------
// Launch: prep (transpose W + zero hist) -> main -> finalize, all on stream.
// Workspace: [0,2560) u32 hist; [4096, 4096+655360) bf16 Wt.
// ---------------------------------------------------------------------------
extern "C" void kernel_launch(void* const* d_in, const int* in_sizes, int n_in,
                              void* d_out, int out_size, void* d_ws,
                              size_t ws_size, hipStream_t stream) {
  (void)in_sizes; (void)n_in; (void)out_size; (void)ws_size;
  const float* hidden = (const float*)d_in[0];   // (16,4096,512) f32
  const float* W      = (const float*)d_in[1];   // (512,640) f32
  const float* bias   = (const float*)d_in[2];   // (640,) f32
  const float* codev  = (const float*)d_in[3];   // (1,640,128) f32
  float* out = (float*)d_out;                    // 16777216 + 1 floats

  unsigned* hist        = (unsigned*)d_ws;
  unsigned short* Wt    = (unsigned short*)((char*)d_ws + 4096);

  vq_prep<<<640, 256, 0, stream>>>(W, Wt, hist);
  vq_main<<<512, 256, 128 * 512 * 2, stream>>>(hidden, Wt, bias, codev, out,
                                               hist);
  vq_finalize<<<1, 256, 0, stream>>>(hist, out + 16777216ull);
}
